// MultiHeadAttention_180388626819
// MI455X (gfx1250) — matrix operand order repaired and checked by reading, not compiled, not run on hardware
//
#include <hip/hip_runtime.h>
#include <hip/hip_bf16.h>
#include <stdint.h>

// ---------------------------------------------------------------------------
// MHA forward for MI455X (gfx1250), bf16 WMMA (f32 accumulate) everywhere.
//   x:[4,2048,1024] Wq:[1024,1024] Wkv:[1024,2048] Wo:[1024,1024] -> out f32
// ---------------------------------------------------------------------------

typedef __bf16 bf16_t;
typedef __attribute__((ext_vector_type(16))) __bf16 v16bf;
typedef __attribute__((ext_vector_type(8)))  __bf16 v8bf;
typedef __attribute__((ext_vector_type(8)))  float  v8f;
typedef __attribute__((ext_vector_type(4)))  int    v4i;

#define HEADS 16
#define DK    64
#define DM    1024
#define BATCH 4
#define SEQ   2048
#define MTOT  (BATCH * SEQ)   // 8192 total rows

#if defined(__AMDGCN__) && __has_builtin(__builtin_amdgcn_global_load_async_to_lds_b128)
#define HAVE_ASYNC 1
#else
#define HAVE_ASYNC 0
#endif

#if defined(__AMDGCN__) && __has_builtin(__builtin_amdgcn_permlane16)
#define HAVE_PERMLANE 1
#else
#define HAVE_PERMLANE 0
#endif

// workspace layout (bytes); AO buffer aliases XB (x_bf16 dead after KV gemm)
static constexpr size_t OFF_XB   = 0;                                   // 16 MiB
static constexpr size_t OFF_WQT  = OFF_XB   + (size_t)16 * 1024 * 1024; //  2 MiB
static constexpr size_t OFF_WKVT = OFF_WQT  + (size_t)2  * 1024 * 1024; //  4 MiB
static constexpr size_t OFF_WOT  = OFF_WKVT + (size_t)4  * 1024 * 1024; //  2 MiB
static constexpr size_t OFF_QB   = OFF_WOT  + (size_t)2  * 1024 * 1024; // 16 MiB
static constexpr size_t OFF_KVB  = OFF_QB   + (size_t)16 * 1024 * 1024; // 32 MiB
static constexpr size_t OFF_VTB  = OFF_KVB  + (size_t)32 * 1024 * 1024; // 16 MiB
// total need: 88 MiB

__device__ __forceinline__ bf16_t to_bf16(float f) {
  union { float f; uint32_t u; } in; in.f = f;
  uint32_t u = in.u;
  u += 0x7FFFu + ((u >> 16) & 1u);          // round-to-nearest-even
  union { uint16_t s; bf16_t b; } out; out.s = (uint16_t)(u >> 16);
  return out.b;
}

__device__ __forceinline__ v8f wmma_bf16(v16bf a, v16bf b, v8f c) {
  return __builtin_amdgcn_wmma_f32_16x16x32_bf16(
      /*neg_a=*/false, a, /*neg_b=*/false, b,
      /*c_mod=*/(short)0, c, /*reuse_a=*/false, /*reuse_b=*/false);
}

// ---- cross-lane xor shuffle within 16-lane rows (VALU permlane16 path) ----
template <int MASK, unsigned S0, unsigned S1>
__device__ __forceinline__ float xshuf(float v) {
#if HAVE_PERMLANE
  return __uint_as_float(__builtin_amdgcn_permlane16(
      __float_as_uint(v), __float_as_uint(v), S0, S1, false, false));
#else
  return __shfl_xor(v, MASK, 32);
#endif
}

__device__ __forceinline__ float rowmax16(float v) {
  v = fmaxf(v, xshuf<1, 0x67452301u, 0xEFCDAB89u>(v));
  v = fmaxf(v, xshuf<2, 0x54761032u, 0xDCFE98BAu>(v));
  v = fmaxf(v, xshuf<4, 0x32107654u, 0xBA98FEDCu>(v));
  v = fmaxf(v, xshuf<8, 0xFEDCBA98u, 0x76543210u>(v));
  return v;
}

__device__ __forceinline__ float rowsum16(float v) {
  v += xshuf<1, 0x67452301u, 0xEFCDAB89u>(v);
  v += xshuf<2, 0x54761032u, 0xDCFE98BAu>(v);
  v += xshuf<4, 0x32107654u, 0xBA98FEDCu>(v);
  v += xshuf<8, 0xFEDCBA98u, 0x76543210u>(v);
  return v;
}

// A fragment: 16(M) x 32(K) bf16, row-major source, two 16B loads per lane.
__device__ __forceinline__ v16bf load_frag_a(const bf16_t* base, int ld,
                                             int m0, int k0) {
  const int lane = threadIdx.x & 31;
  const int row  = m0 + (lane & 15);
  const int koff = (lane < 16) ? 0 : 8;
  const bf16_t* p = base + (size_t)row * ld + k0 + koff;
  v8bf lo = *(const v8bf*)p;
  v8bf hi = *(const v8bf*)(p + 16);
  v16bf r;
#pragma unroll
  for (int i = 0; i < 8; ++i) { r[i] = lo[i]; r[i + 8] = hi[i]; }
  return r;
}

// B fragment: 32(K) x 16(N) bf16 where memory holds Bt[N][K] row-major
// (per-lane fixed N = row of Bt, K = koff..koff+7 and 16+koff..16+koff+7). Two 16B loads per lane.
__device__ __forceinline__ v16bf load_frag_b(const bf16_t* base, int ld,
                                             int n0, int k0) {
  const int lane = threadIdx.x & 31;
  const int row  = n0 + (lane & 15);
  const int koff = (lane < 16) ? 0 : 8;
  const bf16_t* p = base + (size_t)row * ld + k0 + koff;
  v8bf lo = *(const v8bf*)p;
  v8bf hi = *(const v8bf*)(p + 16);
  v16bf r;
#pragma unroll
  for (int i = 0; i < 8; ++i) { r[i] = lo[i]; r[i + 8] = hi[i]; }
  return r;
}

// -------------------------- conversion kernels -----------------------------

__global__ __launch_bounds__(256) void cvt_f32_bf16(const float* __restrict__ src,
                                                    bf16_t* __restrict__ dst, int n) {
  for (int i = blockIdx.x * blockDim.x + threadIdx.x; i < n;
       i += gridDim.x * blockDim.x)
    dst[i] = to_bf16(src[i]);
}

// W[k][n] (fp32, K x Nout) -> Wt[n][k] (bf16, Nout x K)
__global__ __launch_bounds__(256) void wtrans_bf16(const float* __restrict__ W,
                                                   bf16_t* __restrict__ Wt,
                                                   int K, int Nout) {
  const int total = K * Nout;
  for (int i = blockIdx.x * blockDim.x + threadIdx.x; i < total;
       i += gridDim.x * blockDim.x) {
    const int n = i / K, k = i - n * K;
    Wt[i] = to_bf16(W[(size_t)k * Nout + n]);
  }
}

// per-head V transpose: VT[(b*16+h)*64 + d][n] = KV[b*2048+n][1024 + h*64 + d]
__global__ __launch_bounds__(256) void vtrans_bf16(const bf16_t* __restrict__ KVb,
                                                   bf16_t* __restrict__ VTb) {
  const int total = BATCH * HEADS * DK * SEQ;  // 8388608
  for (int i = blockIdx.x * blockDim.x + threadIdx.x; i < total;
       i += gridDim.x * blockDim.x) {
    const int n = i & (SEQ - 1);
    int rest = i >> 11;
    const int d = rest & (DK - 1); rest >>= 6;
    const int h = rest & (HEADS - 1);
    const int b = rest >> 4;
    VTb[i] = KVb[((size_t)(b * SEQ + n)) * (2 * DM) + DM + h * DK + d];
  }
}

// ------------------------------- GEMM --------------------------------------
// C[M][Nn] = A[M][K] * Bt[Nn][K]^T ; block tile 128x128, 8 waves of 32x64.
// Async (gfx1250): 128x32 A/B panels double-buffered in LDS via
// GLOBAL_LOAD_ASYNC_TO_LDS_B128; fragments then come from LDS with 2-4x reuse.

#if HAVE_ASYNC
__device__ __forceinline__ void async_copy16(const bf16_t* gp, bf16_t* lp) {
  __builtin_amdgcn_global_load_async_to_lds_b128(
      (__attribute__((address_space(1))) v4i*)(gp),
      (__attribute__((address_space(3))) v4i*)(lp), 0, 0);
}

// stage one 128x32 bf16 panel (8 KiB) = 512 x 16B chunks, 2 per thread.
__device__ __forceinline__ void stage_panel_async(const bf16_t* g, int ld,
                                                  int row0, int k0,
                                                  bf16_t* lds) {
  const int tid = threadIdx.x;
#pragma unroll
  for (int c = tid; c < 512; c += 256) {
    const int row = c >> 2;
    const int seg = c & 3;
    async_copy16(g + (size_t)(row0 + row) * ld + k0 + seg * 8,
                 lds + row * 32 + seg * 8);
  }
}
#endif

template <bool OUT_BF16>
__global__ __launch_bounds__(256) void gemm_bf16(const bf16_t* __restrict__ A,
                                                 const bf16_t* __restrict__ Bt,
                                                 void* __restrict__ C,
                                                 int M, int Nn, int K) {
  const int wave = threadIdx.x >> 5;
  const int lane = threadIdx.x & 31;
  const int blockM = blockIdx.x * 128;
  const int blockN = blockIdx.y * 128;
  const int wm = (wave >> 1) * 32;   // wave tile origin within block
  const int wn = (wave & 1) * 64;
  (void)M;

  v8f acc[2][4] = {};

#if HAVE_ASYNC
  __shared__ __align__(16) bf16_t Asm[2][128 * 32];
  __shared__ __align__(16) bf16_t Bsm[2][128 * 32];

  stage_panel_async(A, K, blockM, 0, Asm[0]);
  stage_panel_async(Bt, K, blockN, 0, Bsm[0]);

  for (int k0 = 0; k0 < K; k0 += 32) {
    const int buf = (k0 >> 5) & 1;
    if (k0 + 32 < K) {
      stage_panel_async(A, K, blockM, k0 + 32, Asm[buf ^ 1]);
      stage_panel_async(Bt, K, blockN, k0 + 32, Bsm[buf ^ 1]);
      // 4 async instrs/wave/stage, in-order: <=4 outstanding == stage n done
      asm volatile("s_wait_asynccnt 4" ::: "memory");
    } else {
      asm volatile("s_wait_asynccnt 0" ::: "memory");
    }
    __syncthreads();

    v16bf a0 = load_frag_a(&Asm[buf][wm * 32], 32, 0, 0);
    v16bf a1 = load_frag_a(&Asm[buf][(wm + 16) * 32], 32, 0, 0);
    v16bf b[4];
#pragma unroll
    for (int t = 0; t < 4; ++t)
      b[t] = load_frag_b(&Bsm[buf][(wn + t * 16) * 32], 32, 0, 0);
#pragma unroll
    for (int tn = 0; tn < 4; ++tn) {
      acc[0][tn] = wmma_bf16(a0, b[tn], acc[0][tn]);
      acc[1][tn] = wmma_bf16(a1, b[tn], acc[1][tn]);
    }
    __syncthreads();  // protect buf from being restaged next iteration
  }
#else
  for (int k0 = 0; k0 < K; k0 += 32) {
    v16bf a0 = load_frag_a(A, K, blockM + wm, k0);
    v16bf a1 = load_frag_a(A, K, blockM + wm + 16, k0);
    v16bf b[4];
#pragma unroll
    for (int t = 0; t < 4; ++t)
      b[t] = load_frag_b(Bt, K, blockN + wn + t * 16, k0);
#pragma unroll
    for (int tn = 0; tn < 4; ++tn) {
      acc[0][tn] = wmma_bf16(a0, b[tn], acc[0][tn]);
      acc[1][tn] = wmma_bf16(a1, b[tn], acc[1][tn]);
    }
  }
#endif

  const int rhalf = (lane < 16) ? 0 : 8;   // C layout: VGPR v -> M = v + rhalf
  const int cn = lane & 15;
#pragma unroll
  for (int tm = 0; tm < 2; ++tm)
#pragma unroll
    for (int tn = 0; tn < 4; ++tn)
#pragma unroll
      for (int v = 0; v < 8; ++v) {
        const size_t r = (size_t)(blockM + wm + tm * 16 + v + rhalf);
        const size_t c = (size_t)(blockN + wn + tn * 16 + cn);
        if (OUT_BF16) ((bf16_t*)C)[r * Nn + c] = to_bf16(acc[tm][tn][v]);
        else          ((float*)C)[r * Nn + c]  = acc[tm][tn][v];
      }
}

// --------------------------- flash attention -------------------------------
// one wave = 16 query rows of one (b,h); online softmax over 2048 keys in
// 32-key steps; P re-layout (C-tile -> A-frag) via 1KB/wave LDS round trip.
__global__ __launch_bounds__(256) void attn_flash(const bf16_t* __restrict__ Qb,
                                                  const bf16_t* __restrict__ KVb,
                                                  const bf16_t* __restrict__ VTb,
                                                  bf16_t* __restrict__ AOb) {
  __shared__ __align__(16) bf16_t pbuf[8][16 * 32];

  const int wave = threadIdx.x >> 5;
  const int lane = threadIdx.x & 31;
  const int unit = blockIdx.x * 8 + wave;      // 0..8191
  const int mt = unit & 127;                   // 128 row-tiles per (b,h)
  const int bh = unit >> 7;
  const int h = bh & (HEADS - 1);
  const int b = bh >> 4;
  const int m0 = mt * 16;

  const bf16_t* Qh  = Qb  + (size_t)b * SEQ * DM + h * DK;          // ld=DM
  const bf16_t* Kh  = KVb + (size_t)b * SEQ * (2 * DM) + h * DK;    // ld=2*DM
  const bf16_t* VTh = VTb + (size_t)(b * HEADS + h) * DK * SEQ;     // ld=SEQ

  const v16bf qf0 = load_frag_a(Qh, DM, m0, 0);
  const v16bf qf1 = load_frag_a(Qh, DM, m0, 32);

  v8f acc[4] = {};
  float mrun[8], lrun[8];
#pragma unroll
  for (int v = 0; v < 8; ++v) { mrun[v] = -3.0e38f; lrun[v] = 0.f; }

  bf16_t* myp = &pbuf[wave][0];
  const int rhalf = (lane < 16) ? 0 : 8;
  const int cn = lane & 15;

  for (int j0 = 0; j0 < SEQ; j0 += 32) {
    // S = Q * K^T for keys [j0, j0+32): two 16x16 f32 tiles
    v8f s0 = {}, s1 = {};
    s0 = wmma_bf16(qf0, load_frag_b(Kh, 2 * DM, j0, 0), s0);
    s0 = wmma_bf16(qf1, load_frag_b(Kh, 2 * DM, j0, 32), s0);
    s1 = wmma_bf16(qf0, load_frag_b(Kh, 2 * DM, j0 + 16, 0), s1);
    s1 = wmma_bf16(qf1, load_frag_b(Kh, 2 * DM, j0 + 16, 32), s1);

    // online softmax update (row = v + rhalf; reduce across 16-lane row)
#pragma unroll
    for (int v = 0; v < 8; ++v) {
      float a = s0[v] * 0.125f;              // 1/sqrt(64)
      float c = s1[v] * 0.125f;
      const float mx = rowmax16(fmaxf(a, c));
      const float mnew = fmaxf(mrun[v], mx);
      const float p0 = __expf(a - mnew);
      const float p1 = __expf(c - mnew);
      s0[v] = p0; s1[v] = p1;
      const float rs = rowsum16(p0 + p1);
      const float alpha = __expf(mrun[v] - mnew);
      lrun[v] = lrun[v] * alpha + rs;
      mrun[v] = mnew;
      acc[0][v] *= alpha; acc[1][v] *= alpha;
      acc[2][v] *= alpha; acc[3][v] *= alpha;
    }

    // P (16x32 bf16) C-layout -> row-major LDS
#pragma unroll
    for (int v = 0; v < 8; ++v) {
      myp[(v + rhalf) * 32 + cn]      = to_bf16(s0[v]);
      myp[(v + rhalf) * 32 + 16 + cn] = to_bf16(s1[v]);
    }
#if defined(__AMDGCN__)
    asm volatile("s_wait_dscnt 0" ::: "memory");
#endif

    // reload P in A-fragment layout
    v16bf pf;
    {
      const int m = lane & 15;
      const int koff = (lane < 16) ? 0 : 8;
      v8bf lo = *(const v8bf*)&myp[m * 32 + koff];
      v8bf hi = *(const v8bf*)&myp[m * 32 + 16 + koff];
#pragma unroll
      for (int i = 0; i < 8; ++i) { pf[i] = lo[i]; pf[i + 8] = hi[i]; }
    }

    // out += P * V : V fragments from VT (per-lane row = d, contiguous keys)
#pragma unroll
    for (int t = 0; t < 4; ++t)
      acc[t] = wmma_bf16(pf, load_frag_b(VTh, SEQ, t * 16, j0), acc[t]);
  }

  // epilogue: normalize and scatter heads back to [row][h*64+d] (bf16)
#pragma unroll
  for (int t = 0; t < 4; ++t)
#pragma unroll
    for (int v = 0; v < 8; ++v) {
      const float o = acc[t][v] / lrun[v];
      const size_t row = (size_t)(b * SEQ + m0 + v + rhalf);
      AOb[row * DM + h * DK + t * 16 + cn] = to_bf16(o);
    }
}

// ------------------------------ launcher -----------------------------------

extern "C" void kernel_launch(void* const* d_in, const int* in_sizes, int n_in,
                              void* d_out, int out_size, void* d_ws, size_t ws_size,
                              hipStream_t stream) {
  (void)in_sizes; (void)n_in; (void)out_size; (void)ws_size;
  const float* x   = (const float*)d_in[0];
  const float* Wq  = (const float*)d_in[1];
  const float* Wkv = (const float*)d_in[2];
  const float* Wo  = (const float*)d_in[3];
  float* out = (float*)d_out;

  char* ws = (char*)d_ws;
  bf16_t* xb   = (bf16_t*)(ws + OFF_XB);
  bf16_t* WqT  = (bf16_t*)(ws + OFF_WQT);
  bf16_t* WkvT = (bf16_t*)(ws + OFF_WKVT);
  bf16_t* WoT  = (bf16_t*)(ws + OFF_WOT);
  bf16_t* Qb   = (bf16_t*)(ws + OFF_QB);
  bf16_t* KVb  = (bf16_t*)(ws + OFF_KVB);
  bf16_t* VTb  = (bf16_t*)(ws + OFF_VTB);
  bf16_t* AOb  = xb;  // reuse: x_bf16 dead once KV gemm is done

  // 1) precision conversion / weight transposes
  cvt_f32_bf16<<<4096, 256, 0, stream>>>(x, xb, MTOT * DM);
  wtrans_bf16<<<4096, 256, 0, stream>>>(Wq, WqT, DM, DM);
  wtrans_bf16<<<4096, 256, 0, stream>>>(Wkv, WkvT, DM, 2 * DM);
  wtrans_bf16<<<4096, 256, 0, stream>>>(Wo, WoT, DM, DM);

  // 2) projections (bf16 WMMA, f32 accumulate -> bf16)
  gemm_bf16<true><<<dim3(MTOT / 128, DM / 128), 256, 0, stream>>>(
      xb, WqT, Qb, MTOT, DM, DM);
  gemm_bf16<true><<<dim3(MTOT / 128, (2 * DM) / 128), 256, 0, stream>>>(
      xb, WkvT, KVb, MTOT, 2 * DM, DM);

  // 3) per-head V^T so PV fragments are contiguous
  vtrans_bf16<<<8192, 256, 0, stream>>>(KVb, VTb);

  // 4) flash attention (8192 wave-units, 8 waves/block)
  attn_flash<<<1024, 256, 0, stream>>>(Qb, KVb, VTb, AOb);

  // 5) output projection -> fp32
  gemm_bf16<false><<<dim3(MTOT / 128, DM / 128), 256, 0, stream>>>(
      AOb, WoT, out, MTOT, DM, DM);
}
